// FRNN_old_7344394076730
// MI455X (gfx1250) — matrix operand order repaired and checked by reading, not compiled, not run on hardware
//
#include <hip/hip_runtime.h>
#include <hip/hip_bf16.h>
#include <math.h>

// Problem constants (match reference)
#define VIS   2048
#define HID   6144
#define FULLN 8192            // VIS + HID
#define TSTEPS 1024

// WMMA fragment types (CDNA5 gfx1250, wave32)
typedef __attribute__((ext_vector_type(16))) _Float16 v16h;
typedef __attribute__((ext_vector_type(8)))  float    v8f;

// ---------------------------------------------------------------------------
// Kernel 1: convert W (fp32, row-major FULLxFULL) into fp16, pre-swizzled into
// a 16-bit A-matrix fragment layout with contiguous K order (matches B in frnn_step):
//   A tile = 16 rows x 32 K.  lane L (L%16 = row), 16 halves per lane:
//     half i (0..15) -> K = i + (L>=16 ? 16 : 0)
// Storage: Wf[((tile*256 + kchunk)*32 + lane)*16 + i]
// One thread per (tile,kchunk,lane) -> writes 16 contiguous halves.
// ---------------------------------------------------------------------------
__global__ void frnn_convert_w(const float* __restrict__ W,
                               _Float16* __restrict__ Wf) {
    size_t fid = (size_t)blockIdx.x * blockDim.x + threadIdx.x; // 0 .. 512*256*32-1
    int    L   = (int)(fid & 31);
    size_t mc  = fid >> 5;            // tile*256 + kchunk
    int    c   = (int)(mc & 255);
    int    m   = (int)(mc >> 8);
    int    row = m * 16 + (L & 15);
    int    cb  = c * 32 + ((L >= 16) ? 16 : 0);
    const float* src = W + (size_t)row * FULLN + cb;
    _Float16*    dst = Wf + fid * 16;
#pragma unroll
    for (int i = 0; i < 8; ++i) dst[i]     = (_Float16)src[i];
#pragma unroll
    for (int i = 0; i < 8; ++i) dst[8 + i] = (_Float16)src[8 + i];
}

// Zero the initial recurrent state r0 (fp16).
__global__ void frnn_zero_r(_Float16* __restrict__ r) {
    int i = blockIdx.x * blockDim.x + threadIdx.x;
    if (i < FULLN) r[i] = (_Float16)0.0f;
}

// ---------------------------------------------------------------------------
// Kernel 2: one recurrence step.
//   y = W @ r          (WMMA GEMV: B = r replicated across all 16 columns)
//   u = Lam*(y + b) + (1-Lam)*x_t ; r_out = tanh(u) ; out[:VIS] = u[:VIS]
// Grid: 512 blocks (one 16-row tile each), 256 threads = 8 waves.
// Wave w covers K range [w*1024, w*1024+1024) -> 32 WMMAs of K=32.
// ---------------------------------------------------------------------------
__global__ void __launch_bounds__(256)
frnn_step(const _Float16* __restrict__ Wf,
          const _Float16* __restrict__ rin,
          _Float16* __restrict__ rout,
          const float* __restrict__ X,
          const float* __restrict__ bias,
          const float* __restrict__ lam,
          float* __restrict__ out,
          int t) {
    __shared__ float red[8 * 16];

    const int tile = blockIdx.x;          // 0..511
    const int wave = threadIdx.x >> 5;    // 0..7
    const int lane = threadIdx.x & 31;

    // A fragments: contiguous per (tile, kchunk, lane)
    const _Float16* aptr = Wf + ((size_t)tile * 256 + (size_t)wave * 32) * 512
                              + (size_t)lane * 16;
    // B fragments: r replicated in all 16 columns.
    // B 32x16 layout: lanes 0-15 hold K=0..15, lanes 16-31 hold K=16..31,
    // each lane 16 contiguous halves in K.
    const _Float16* bptr = rin + wave * 1024 + ((lane >= 16) ? 16 : 0);

    v8f acc = {};
#pragma unroll 4
    for (int c = 0; c < 32; ++c) {
        v16h a = *(const v16h*)aptr;
        v16h b = *(const v16h*)bptr;
        acc = __builtin_amdgcn_wmma_f32_16x16x32_f16(
            /*neg_a=*/false, a, /*neg_b=*/false, b,
            /*c_mod=*/(short)0, acc, /*reuse_a=*/false, /*reuse_b=*/false);
        aptr += 512;   // next 16x32 fragment (1024 B / wave)
        bptr += 32;    // next 32 K-values of r
    }

    // D layout: lane n (0-15) = column n; VGPR g holds row g (lanes 0-15)
    // or row g+8 (lanes 16-31). Columns are identical -> read lanes 0 and 16.
    if (lane == 0 || lane == 16) {
        const int half = (lane == 16) ? 8 : 0;
#pragma unroll
        for (int g = 0; g < 8; ++g) red[wave * 16 + half + g] = acc[g];
    }
    __syncthreads();

    if (threadIdx.x < 16) {
        const int grow = tile * 16 + threadIdx.x;
        float s = 0.0f;
#pragma unroll
        for (int w = 0; w < 8; ++w) s += red[w * 16 + threadIdx.x];
        const float L  = lam[grow];
        const float xv = (grow < VIS) ? X[(size_t)t * VIS + grow] : 0.0f;
        const float u  = L * (s + bias[grow]) + (1.0f - L) * xv;
        rout[grow] = (_Float16)tanhf(u);
        if (grow < VIS) out[grow] = u;   // last step's write is the answer
    }
}

// ---------------------------------------------------------------------------
// Launch: convert W -> fp16 fragments (128 MB, fits MI455X 192 MB L2 so all
// 1024 GEMV passes after the first are L2-resident instead of 128 GB of HBM
// traffic), zero r0, then 1024 sequential step kernels (graph-capturable).
// Workspace layout: [Wf: FULLN*FULLN fp16 = 128 MB][rA: FULLN fp16][rB: ...]
// ---------------------------------------------------------------------------
extern "C" void kernel_launch(void* const* d_in, const int* in_sizes, int n_in,
                              void* d_out, int out_size, void* d_ws, size_t ws_size,
                              hipStream_t stream) {
    const float* X   = (const float*)d_in[0];   // [T, VIS]
    const float* W   = (const float*)d_in[1];   // [FULLN, FULLN]
    const float* b   = (const float*)d_in[2];   // [FULLN]
    const float* Lam = (const float*)d_in[3];   // [FULLN]
    float*       out = (float*)d_out;           // [VIS]

    _Float16* Wf = (_Float16*)d_ws;
    _Float16* rA = (_Float16*)((char*)d_ws + (size_t)FULLN * FULLN * sizeof(_Float16));
    _Float16* rB = rA + FULLN;

    // 512 tiles * 256 kchunks * 32 lanes threads, 256/block
    frnn_convert_w<<<16384, 256, 0, stream>>>(W, Wf);
    frnn_zero_r<<<(FULLN + 255) / 256, 256, 0, stream>>>(rA);

    for (int t = 0; t < TSTEPS; ++t) {
        const _Float16* rin  = (t & 1) ? rB : rA;
        _Float16*       rout = (t & 1) ? rA : rB;
        frnn_step<<<512, 256, 0, stream>>>(Wf, rin, rout, X, b, Lam, out, t);
    }
}
